// DecoderRNN_85658827751976
// MI455X (gfx1250) — matrix operand order repaired and checked by reading, not compiled, not run on hardware
//
#include <hip/hip_runtime.h>
#include <hip/hip_bf16.h>
#include <math.h>

#define B_   128
#define T_   28
#define TS_  27
#define H_   512
#define DW_  512
#define P_   256
#define NCH_ 16
#define CH_  32
#define G_   2080
#define SV_  40
#define SP_  28
#define V_   30000

typedef __bf16 bf16_t;
typedef bf16_t v16bf __attribute__((ext_vector_type(16)));
typedef bf16_t v8bf  __attribute__((ext_vector_type(8)));
typedef float  v8f   __attribute__((ext_vector_type(8)));

__device__ __forceinline__ float sigmoidf_(float x) { return 1.0f / (1.0f + expf(-x)); }

// ---------------------------------------------------------------------------
// Plain f32 -> bf16 copy (for encoder activations used as GEMM A-matrices)
// ---------------------------------------------------------------------------
__global__ void cvt_bf16_kernel(const float* __restrict__ src, bf16_t* __restrict__ dst, size_t n) {
    size_t i = (size_t)blockIdx.x * blockDim.x + threadIdx.x;
    size_t stride = (size_t)gridDim.x * blockDim.x;
    for (; i < n; i += stride) dst[i] = (bf16_t)src[i];
}

// ---------------------------------------------------------------------------
// Tiled transpose + f32->bf16: src[K,N] (row-major f32) -> dst[N,K] (bf16).
// ---------------------------------------------------------------------------
__global__ void cvt_transpose_kernel(const float* __restrict__ src, bf16_t* __restrict__ dst,
                                     int K, int N) {
    __shared__ float tile[32][33];
    const int kb = blockIdx.y * 32;
    const int nb = blockIdx.x * 32;
    const int tx = threadIdx.x;          // 0..31
    const int ty = threadIdx.y;          // 0..7
    for (int i = ty; i < 32; i += 8) {
        int k = kb + i, n = nb + tx;
        if (k < K && n < N) tile[i][tx] = src[(size_t)k * N + n];
    }
    __syncthreads();
    for (int i = ty; i < 32; i += 8) {
        int n = nb + i, k = kb + tx;
        if (n < N && k < K) dst[(size_t)n * K + k] = (bf16_t)tile[tx][i];
    }
}

// ---------------------------------------------------------------------------
// Embedding gather: xs[t][b][:] = embedding[targets[b][t]] for t in [0,27)
// ---------------------------------------------------------------------------
__global__ void embed_kernel(const int* __restrict__ targets, const float* __restrict__ emb,
                             float* __restrict__ xs) {
    int idx = blockIdx.x;           // TS_*B_ blocks
    int t = idx / B_, b = idx % B_;
    int tok = targets[b * T_ + t];
    const float* src = emb + (size_t)tok * DW_;
    float* dst = xs + ((size_t)t * B_ + b) * DW_;
    for (int d = threadIdx.x; d < DW_; d += blockDim.x) dst[d] = src[d];
}

// ---------------------------------------------------------------------------
// out_bf[b][:] = bf16(concat(a[b][:da], c[b][:dc]))   (GEMM A-operand feed)
// ---------------------------------------------------------------------------
__global__ void concat2_bf16_kernel(const float* __restrict__ a, int da,
                                    const float* __restrict__ c, int dc,
                                    bf16_t* __restrict__ out) {
    int b = blockIdx.x;
    int D = da + dc;
    for (int i = threadIdx.x; i < D; i += blockDim.x)
        out[(size_t)b * D + i] = (bf16_t)((i < da) ? a[(size_t)b * da + i]
                                                   : c[(size_t)b * dc + (i - da)]);
}

// ---------------------------------------------------------------------------
// bf16 WMMA GEMM, software-pipelined, LDA compile-time:
//   C[M,N](f32) (+)= A[M,K](bf16,row-major,lda=LDA) * W([N,ldwt] bf16, transposed) + bias
//   - wave computes 64(M) x 16(N): 4 f32 accumulators, B-fragment reused 4x
//   - all 8 A-loads use ONE base pointer + constant offsets (s*16*LDA folds
//     into the 24-bit instruction offset -> no per-subtile pointers, no spills)
//   - next iteration's loads issued before current WMMAs (latency hiding)
//   - out_tb: GEMM row r = t*128+b stored at output row (b*27 + t)
// ---------------------------------------------------------------------------
template <int LDA>
__global__ void gemm_bf16_wmma(const bf16_t* __restrict__ A,
                               const bf16_t* __restrict__ Wt, int ldwt,
                               const float* __restrict__ bias,
                               float* __restrict__ C, int ldc,
                               int M, int N, int K,
                               int accumulate, int out_tb) {
    const int lane  = threadIdx.x & 31;
    const int wave  = threadIdx.x >> 5;
    const int half  = lane >> 4;       // 0 or 1
    const int l16   = lane & 15;
    const int mbase = blockIdx.x * 64;
    const int ntile = blockIdx.y * 4 + wave;
    if (ntile * 16 >= N) return;       // wave-uniform exit; EXEC all-1 for WMMA
    const int n = ntile * 16 + l16;

    v8f acc[4];
    if (accumulate) {
#pragma unroll
        for (int s = 0; s < 4; ++s)
#pragma unroll
            for (int r = 0; r < 8; ++r) {
                int m = mbase + s * 16 + half * 8 + r;
                acc[s][r] = C[(size_t)m * ldc + n];
            }
    } else {
#pragma unroll
        for (int s = 0; s < 4; ++s)
#pragma unroll
            for (int r = 0; r < 8; ++r) acc[s][r] = 0.0f;
    }

    const bf16_t* Wrow  = Wt + (size_t)n * ldwt + half * 16;            // lane's weight column
    const bf16_t* Abase = A + (size_t)(mbase + l16) * LDA + half * 16;  // single A base

    // helper: A-fragment for subtile s at k-offset k0 (all offsets constant-folded)
    auto load_a = [&](int s, int k0) -> v16bf {
        const bf16_t* p = Abase + s * 16 * LDA + k0;
        v8bf lo = *(const v8bf*)(p);
        v8bf hi = *(const v8bf*)(p + 8);
        v16bf f;
#pragma unroll
        for (int e = 0; e < 8; ++e) { f[e] = lo[e]; f[8 + e] = hi[e]; }
        return f;
    };

    // prologue: k0 = 0 fragments
    v16bf bcur = *(const v16bf*)(Wrow);
    v16bf acur[4];
#pragma unroll
    for (int s = 0; s < 4; ++s) acur[s] = load_a(s, 0);

    int k0 = 0;
    for (; k0 + 32 < K; k0 += 32) {
        // issue next iteration's loads first (overlap with WMMAs below)
        v16bf bnxt = *(const v16bf*)(Wrow + k0 + 32);
        v16bf anxt[4];
#pragma unroll
        for (int s = 0; s < 4; ++s) anxt[s] = load_a(s, k0 + 32);

        if (k0 + 64 < K)   // gfx1250 global_prefetch_b8 two slabs ahead
            __builtin_prefetch(Wrow + k0 + 64, 0, 1);

#pragma unroll
        for (int s = 0; s < 4; ++s)
            acc[s] = __builtin_amdgcn_wmma_f32_16x16x32_bf16(
                false, acur[s], false, bcur, (short)0, acc[s], false, false);

        bcur = bnxt;
#pragma unroll
        for (int s = 0; s < 4; ++s) acur[s] = anxt[s];
    }
    // epilogue
#pragma unroll
    for (int s = 0; s < 4; ++s)
        acc[s] = __builtin_amdgcn_wmma_f32_16x16x32_bf16(
            false, acur[s], false, bcur, (short)0, acc[s], false, false);

    const float bv = bias ? bias[n] : 0.0f;
#pragma unroll
    for (int s = 0; s < 4; ++s)
#pragma unroll
        for (int r = 0; r < 8; ++r) {
            int m = mbase + s * 16 + half * 8 + r;
            float val = acc[s][r] + bv;
            long row = out_tb ? ((long)(m & (B_ - 1)) * TS_ + (m >> 7)) : (long)m;
            C[row * (long)ldc + n] = val;
        }
}

// ---------------------------------------------------------------------------
// Additive attention: e[b,s] = w2 . tanh(pre[b,s,:] + hidproj[b,:])
//                     a = softmax_s(e);  ctx[b,:] = sum_s a[s]*enc[b,s,:]
// Optionally also writes a bf16 copy of ctx (GEMM A-operand feed).
// ---------------------------------------------------------------------------
__global__ void attend_kernel(const float* __restrict__ pre, const float* __restrict__ hidproj,
                              const float* __restrict__ w2, const float* __restrict__ enc,
                              float* __restrict__ ctx, bf16_t* __restrict__ ctx_bf,
                              int S, int D) {
    const int b = blockIdx.x;
    __shared__ float sc[64];
    const int lane = threadIdx.x & 31;
    const int wave = threadIdx.x >> 5;           // 256 threads -> 8 waves

    const float* hp = hidproj + (size_t)b * D;
    for (int s = wave; s < S; s += 8) {
        const float* pr = pre + ((size_t)b * S + s) * D;
        float acc = 0.0f;
        for (int d = lane; d < D; d += 32)
            acc += w2[d] * tanhf(pr[d] + hp[d]);
        for (int off = 16; off > 0; off >>= 1) acc += __shfl_down(acc, off, 32);
        if (lane == 0) sc[s] = acc;
    }
    __syncthreads();
    if (threadIdx.x == 0) {                      // softmax over S (<=40)
        float mx = sc[0];
        for (int s = 1; s < S; ++s) mx = fmaxf(mx, sc[s]);
        float sum = 0.0f;
        for (int s = 0; s < S; ++s) { sc[s] = expf(sc[s] - mx); sum += sc[s]; }
        float inv = 1.0f / sum;
        for (int s = 0; s < S; ++s) sc[s] *= inv;
    }
    __syncthreads();
    for (int d = threadIdx.x; d < D; d += blockDim.x) {
        float acc = 0.0f;
        for (int s = 0; s < S; ++s) acc += sc[s] * enc[((size_t)b * S + s) * D + d];
        ctx[(size_t)b * D + d] = acc;
        if (ctx_bf) ctx_bf[(size_t)b * D + d] = (bf16_t)acc;
    }
}

// ---------------------------------------------------------------------------
// ON-LSTM elementwise; writes f32 state + bf16 copies used as GEMM A-operands.
// ---------------------------------------------------------------------------
__global__ void onlstm_elem_kernel(const float* __restrict__ gts, const float* __restrict__ c_in,
                                   float* __restrict__ h_out, bf16_t* __restrict__ h_out_bf,
                                   float* __restrict__ c_out,
                                   bf16_t* __restrict__ h1all_bf, int t) {
    const int b = blockIdx.x;
    __shared__ float ci[NCH_], cf[NCH_];
    const float* g = gts + (size_t)b * G_;
    if (threadIdx.x == 0) {                      // cingate = 1 - cumsum(softmax(master[:16]))
        float mx = g[0];
        for (int i = 1; i < NCH_; ++i) mx = fmaxf(mx, g[i]);
        float e[NCH_], sum = 0.0f;
        for (int i = 0; i < NCH_; ++i) { e[i] = expf(g[i] - mx); sum += e[i]; }
        float cum = 0.0f;
        for (int i = 0; i < NCH_; ++i) { cum += e[i] / sum; ci[i] = 1.0f - cum; }
    }
    if (threadIdx.x == 1) {                      // cforget = cumsum(softmax(master[16:32]))
        float mx = g[NCH_];
        for (int i = 1; i < NCH_; ++i) mx = fmaxf(mx, g[NCH_ + i]);
        float e[NCH_], sum = 0.0f;
        for (int i = 0; i < NCH_; ++i) { e[i] = expf(g[NCH_ + i] - mx); sum += e[i]; }
        float cum = 0.0f;
        for (int i = 0; i < NCH_; ++i) { cum += e[i] / sum; cf[i] = cum; }
    }
    __syncthreads();
    const int tid = threadIdx.x;                 // 512 threads
    const int nch = tid >> 5, ch = tid & 31;
    const float* rest = g + 2 * NCH_;
    float og = sigmoidf_(rest[(0 * NCH_ + nch) * CH_ + ch]);
    float cg = tanhf(rest[(1 * NCH_ + nch) * CH_ + ch]);
    float ig = sigmoidf_(rest[(2 * NCH_ + nch) * CH_ + ch]);
    float fg = sigmoidf_(rest[(3 * NCH_ + nch) * CH_ + ch]);
    float cih = ci[nch], cfh = cf[nch];
    float ov = cfh * cih;
    float fgate = fg * ov + (cfh - ov);
    float igate = ig * ov + (cih - ov);
    float cy = fgate * c_in[(size_t)b * H_ + tid] + igate * cg;
    float hy = og * tanhf(cy);
    c_out[(size_t)b * H_ + tid] = cy;
    h_out[(size_t)b * H_ + tid] = hy;
    h_out_bf[(size_t)b * H_ + tid] = (bf16_t)hy;
    if (h1all_bf) h1all_bf[((size_t)t * B_ + b) * H_ + tid] = (bf16_t)hy;
}

// ---------------------------------------------------------------------------
// In-place log_softmax over last dim (row = one (b,t) pair, V_ = 30000)
// ---------------------------------------------------------------------------
__global__ void logsoftmax_kernel(float* __restrict__ out) {
    float* row = out + (size_t)blockIdx.x * V_;
    __shared__ float red[256];
    float mx = -INFINITY;
    for (int i = threadIdx.x; i < V_; i += 256) mx = fmaxf(mx, row[i]);
    red[threadIdx.x] = mx; __syncthreads();
    for (int s = 128; s > 0; s >>= 1) {
        if (threadIdx.x < s) red[threadIdx.x] = fmaxf(red[threadIdx.x], red[threadIdx.x + s]);
        __syncthreads();
    }
    mx = red[0]; __syncthreads();
    float sum = 0.0f;
    for (int i = threadIdx.x; i < V_; i += 256) sum += expf(row[i] - mx);
    red[threadIdx.x] = sum; __syncthreads();
    for (int s = 128; s > 0; s >>= 1) {
        if (threadIdx.x < s) red[threadIdx.x] += red[threadIdx.x + s];
        __syncthreads();
    }
    float lse = logf(red[0]) + mx;
    for (int i = threadIdx.x; i < V_; i += 256) row[i] = row[i] - lse;
}

// ---------------------------------------------------------------------------
// Host orchestration
// ---------------------------------------------------------------------------
static inline dim3 gemm_grid(int M, int N) {
    return dim3(M / 64, (N / 16 + 3) / 4);
}

extern "C" void kernel_launch(void* const* d_in, const int* in_sizes, int n_in,
                              void* d_out, int out_size, void* d_ws, size_t ws_size,
                              hipStream_t stream) {
    (void)in_sizes; (void)n_in; (void)out_size; (void)ws_size;
    const float* enc_v     = (const float*)d_in[0];
    const float* enc_p     = (const float*)d_in[2];
    const int*   targets   = (const int*)d_in[4];
    const float* embedding = (const float*)d_in[5];
    const float* av_W1 = (const float*)d_in[6];
    const float* av_b1 = (const float*)d_in[7];
    const float* av_w2 = (const float*)d_in[8];
    const float* ap_W1 = (const float*)d_in[9];
    const float* ap_b1 = (const float*)d_in[10];
    const float* ap_w2 = (const float*)d_in[11];
    const float* ih_W0 = (const float*)d_in[12];
    const float* ih_b0 = (const float*)d_in[13];
    const float* hh_W0 = (const float*)d_in[14];
    const float* hh_b0 = (const float*)d_in[15];
    const float* ph_W0 = (const float*)d_in[16];
    const float* ph_b0 = (const float*)d_in[17];
    const float* ih_W1 = (const float*)d_in[18];
    const float* ih_b1 = (const float*)d_in[19];
    const float* hh_W1 = (const float*)d_in[20];
    const float* hh_b1 = (const float*)d_in[21];
    const float* ph_W1 = (const float*)d_in[22];
    const float* ph_b1 = (const float*)d_in[23];
    const float* out_W = (const float*)d_in[24];
    const float* out_b = (const float*)d_in[25];
    float* out = (float*)d_out;

    // ---- workspace carve-up -------------------------------------------------
    char* cur = (char*)d_ws;
    auto alloc = [&](size_t bytes) -> void* {
        void* p = (void*)cur;
        cur += (bytes + 255) & ~(size_t)255;
        return p;
    };
    // transposed bf16 weights: [N, K] layout
    bf16_t* Wav  = (bf16_t*)alloc(sizeof(bf16_t) * 512 * 1536);
    bf16_t* Wap  = (bf16_t*)alloc(sizeof(bf16_t) * 256 * 1280);
    bf16_t* Wih0 = (bf16_t*)alloc(sizeof(bf16_t) * G_  * 1024);
    bf16_t* Whh0 = (bf16_t*)alloc(sizeof(bf16_t) * G_  * 512);
    bf16_t* Wih1 = (bf16_t*)alloc(sizeof(bf16_t) * G_  * 512);
    bf16_t* Whh1 = (bf16_t*)alloc(sizeof(bf16_t) * G_  * 512);
    bf16_t* Wph0 = (bf16_t*)alloc(sizeof(bf16_t) * 32  * 256);
    bf16_t* Wph1 = (bf16_t*)alloc(sizeof(bf16_t) * 32  * 256);
    bf16_t* Wout = (bf16_t*)alloc(sizeof(bf16_t) * (size_t)V_ * 512);

    // bf16 activations (GEMM A-operands)
    bf16_t* enc_v_bf = (bf16_t*)alloc(sizeof(bf16_t) * B_ * SV_ * H_);
    bf16_t* enc_p_bf = (bf16_t*)alloc(sizeof(bf16_t) * B_ * SP_ * P_);
    bf16_t* ah_bf    = (bf16_t*)alloc(sizeof(bf16_t) * B_ * 2 * H_);
    bf16_t* xbuf_bf  = (bf16_t*)alloc(sizeof(bf16_t) * B_ * 2 * H_);
    bf16_t* cp_bf    = (bf16_t*)alloc(sizeof(bf16_t) * B_ * P_);
    bf16_t* h0a_bf   = (bf16_t*)alloc(sizeof(bf16_t) * B_ * H_);
    bf16_t* h0b_bf   = (bf16_t*)alloc(sizeof(bf16_t) * B_ * H_);
    bf16_t* h1a_bf   = (bf16_t*)alloc(sizeof(bf16_t) * B_ * H_);
    bf16_t* h1b_bf   = (bf16_t*)alloc(sizeof(bf16_t) * B_ * H_);
    bf16_t* H1all_bf = (bf16_t*)alloc(sizeof(bf16_t) * TS_ * B_ * H_);

    float* xs    = (float*)alloc(sizeof(float) * TS_ * B_ * DW_);
    float* encvp = (float*)alloc(sizeof(float) * B_ * SV_ * H_);
    float* encpp = (float*)alloc(sizeof(float) * B_ * SP_ * P_);
    float* h0a = (float*)alloc(sizeof(float) * B_ * H_);
    float* h0b = (float*)alloc(sizeof(float) * B_ * H_);
    float* c0a = (float*)alloc(sizeof(float) * B_ * H_);
    float* c0b = (float*)alloc(sizeof(float) * B_ * H_);
    float* h1a = (float*)alloc(sizeof(float) * B_ * H_);
    float* h1b = (float*)alloc(sizeof(float) * B_ * H_);
    float* c1a = (float*)alloc(sizeof(float) * B_ * H_);
    float* c1b = (float*)alloc(sizeof(float) * B_ * H_);
    float* avhid = (float*)alloc(sizeof(float) * B_ * H_);
    float* aphid = (float*)alloc(sizeof(float) * B_ * P_);
    float* cv    = (float*)alloc(sizeof(float) * B_ * H_);
    float* cp    = (float*)alloc(sizeof(float) * B_ * P_);
    float* gts   = (float*)alloc(sizeof(float) * B_ * G_);

    // ---- weight transpose + bf16 conversion (coalesced, LDS-tiled) ---------
    dim3 tb(32, 8);
    auto cvtT = [&](const float* s, bf16_t* d, int K, int N) {
        dim3 g((N + 31) / 32, (K + 31) / 32);
        cvt_transpose_kernel<<<g, tb, 0, stream>>>(s, d, K, N);
    };
    cvtT(av_W1, Wav,  1536, 512);
    cvtT(ap_W1, Wap,  1280, 256);
    cvtT(ih_W0, Wih0, 1024, G_);
    cvtT(hh_W0, Whh0, 512,  G_);
    cvtT(ih_W1, Wih1, 512,  G_);
    cvtT(hh_W1, Whh1, 512,  G_);
    cvtT(ph_W0, Wph0, 256,  32);
    cvtT(ph_W1, Wph1, 256,  32);
    cvtT(out_W, Wout, 512,  V_);

    // encoder activations -> bf16 (A-operands of the precompute GEMMs)
    cvt_bf16_kernel<<<2048, 256, 0, stream>>>(enc_v, enc_v_bf, (size_t)B_ * SV_ * H_);
    cvt_bf16_kernel<<<2048, 256, 0, stream>>>(enc_p, enc_p_bf, (size_t)B_ * SP_ * P_);

    // ---- embedding gather ---------------------------------------------------
    embed_kernel<<<TS_ * B_, 256, 0, stream>>>(targets, embedding, xs);

    // ---- step-invariant attention pre-projections ---------------------------
    gemm_bf16_wmma<H_><<<gemm_grid(B_ * SV_, H_), 128, 0, stream>>>(
        enc_v_bf, Wav, 1536, av_b1, encvp, H_, B_ * SV_, H_, H_, 0, 0);
    gemm_bf16_wmma<P_><<<gemm_grid(B_ * SP_, P_), 128, 0, stream>>>(
        enc_p_bf, Wap, 1280, ap_b1, encpp, P_, B_ * SP_, P_, P_, 0, 0);

    // ---- zero initial states ------------------------------------------------
    hipMemsetAsync(h0a, 0, sizeof(float) * B_ * H_, stream);
    hipMemsetAsync(c0a, 0, sizeof(float) * B_ * H_, stream);
    hipMemsetAsync(h1a, 0, sizeof(float) * B_ * H_, stream);
    hipMemsetAsync(c1a, 0, sizeof(float) * B_ * H_, stream);
    hipMemsetAsync(h0a_bf, 0, sizeof(bf16_t) * B_ * H_, stream);
    hipMemsetAsync(h1a_bf, 0, sizeof(bf16_t) * B_ * H_, stream);

    float  *h0c = h0a, *c0c = c0a, *h1c = h1a, *c1c = c1a;
    float  *h0n = h0b, *c0n = c0b, *h1n = h1b, *c1n = c1b;
    bf16_t *h0c_bf = h0a_bf, *h1c_bf = h1a_bf;
    bf16_t *h0n_bf = h0b_bf, *h1n_bf = h1b_bf;

    for (int t = 0; t < TS_; ++t) {
        // ah = bf16(concat(h0, h1))
        concat2_bf16_kernel<<<B_, 256, 0, stream>>>(h0c, H_, h1c, H_, ah_bf);
        // avhid = ah @ av_W1[512:1536,:]  (Wt k-offset 512)
        gemm_bf16_wmma<2 * H_><<<gemm_grid(B_, H_), 128, 0, stream>>>(
            ah_bf, Wav + 512, 1536, nullptr, avhid, H_, B_, H_, 2 * H_, 0, 0);
        // aphid = ah @ ap_W1[256:1280,:]  (Wt k-offset 256)
        gemm_bf16_wmma<2 * H_><<<gemm_grid(B_, P_), 128, 0, stream>>>(
            ah_bf, Wap + 256, 1280, nullptr, aphid, P_, B_, P_, 2 * H_, 0, 0);
        // attention contexts (cp also emitted in bf16 for the ph GEMMs)
        attend_kernel<<<B_, 256, 0, stream>>>(encvp, avhid, av_w2, enc_v, cv, nullptr, SV_, H_);
        attend_kernel<<<B_, 256, 0, stream>>>(encpp, aphid, ap_w2, enc_p, cp, cp_bf, SP_, P_);
        // x = bf16(concat(xt, cv))
        concat2_bf16_kernel<<<B_, 256, 0, stream>>>(xs + (size_t)t * B_ * DW_, DW_, cv, H_, xbuf_bf);

        // ---- cell 0
        gemm_bf16_wmma<2 * H_><<<gemm_grid(B_, G_), 128, 0, stream>>>(
            xbuf_bf, Wih0, 1024, ih_b0, gts, G_, B_, G_, 2 * H_, 0, 0);
        gemm_bf16_wmma<H_><<<gemm_grid(B_, G_), 128, 0, stream>>>(
            h0c_bf, Whh0, 512, hh_b0, gts, G_, B_, G_, H_, 1, 0);
        gemm_bf16_wmma<P_><<<gemm_grid(B_, 32), 128, 0, stream>>>(
            cp_bf, Wph0, 256, ph_b0, gts, G_, B_, 32, P_, 1, 0);
        onlstm_elem_kernel<<<B_, 512, 0, stream>>>(gts, c0c, h0n, h0n_bf, c0n, nullptr, 0);

        // ---- cell 1
        gemm_bf16_wmma<H_><<<gemm_grid(B_, G_), 128, 0, stream>>>(
            h0n_bf, Wih1, 512, ih_b1, gts, G_, B_, G_, H_, 0, 0);
        gemm_bf16_wmma<H_><<<gemm_grid(B_, G_), 128, 0, stream>>>(
            h1c_bf, Whh1, 512, hh_b1, gts, G_, B_, G_, H_, 1, 0);
        gemm_bf16_wmma<P_><<<gemm_grid(B_, 32), 128, 0, stream>>>(
            cp_bf, Wph1, 256, ph_b1, gts, G_, B_, 32, P_, 1, 0);
        onlstm_elem_kernel<<<B_, 512, 0, stream>>>(gts, c1c, h1n, h1n_bf, c1n, H1all_bf, t);

        // swap state buffers
        float* tmp;
        tmp = h0c; h0c = h0n; h0n = tmp;
        tmp = c0c; c0c = c0n; c0n = tmp;
        tmp = h1c; h1c = h1n; h1n = tmp;
        tmp = c1c; c1c = c1n; c1n = tmp;
        bf16_t* tb_;
        tb_ = h0c_bf; h0c_bf = h0n_bf; h0n_bf = tb_;
        tb_ = h1c_bf; h1c_bf = h1n_bf; h1n_bf = tb_;
    }

    // ---- batched vocab projection: (27*128, 512) @ (512, 30000) + out_b ----
    gemm_bf16_wmma<H_><<<gemm_grid(TS_ * B_, V_), 128, 0, stream>>>(
        H1all_bf, Wout, 512, out_b, out, V_, TS_ * B_, V_, H_, 0, 1);

    // ---- in-place log_softmax over vocab ------------------------------------
    logsoftmax_kernel<<<TS_ * B_, 256, 0, stream>>>(out);
}
